// Group_query_Attention_28372553957596
// MI455X (gfx1250) — matrix operand order repaired and checked by reading, not compiled, not run on hardware
//
#include <hip/hip_runtime.h>
#include <hip/hip_bf16.h>

// ---------------------------------------------------------------------------
// GQA forward for MI455X (gfx1250): bf16 WMMA everywhere, fp32 accumulation,
// async global->LDS staging (ASYNCcnt) for attention K/V tiles, DPP16
// row reductions for the online softmax.
// Shapes fixed by setup_inputs(): B=1, T=2048, C=2048, Hq=32, Hkv=8, D=64.
// ---------------------------------------------------------------------------

typedef __bf16 bf16_t;
typedef __attribute__((ext_vector_type(16))) __bf16 v16bf;
typedef __attribute__((ext_vector_type(8)))  float  v8f;

static __device__ __forceinline__ v8f wmma_bf16(v16bf a, v16bf b, v8f c) {
  // D = A(16x32 bf16) * B(32x16 bf16) + C(16x16 f32)
  return __builtin_amdgcn_wmma_f32_16x16x32_bf16(
      /*neg_a=*/false, a, /*neg_b=*/false, b,
      /*c_mod=*/(short)0, c, /*reuse_a=*/false, /*reuse_b=*/false);
}

static __device__ __forceinline__ uint32_t pack2_bf16(float x, float y) {
  union { bf16_t h[2]; uint32_t u; } p;
  p.h[0] = (bf16_t)x;
  p.h[1] = (bf16_t)y;
  return p.u;
}

// Overloads: pack two fp32 -> b32 of two bf16, or pass through two bf16.
static __device__ __forceinline__ uint32_t load_pair_as_bf16x2(const float* p) {
  return pack2_bf16(p[0], p[1]);
}
static __device__ __forceinline__ uint32_t load_pair_as_bf16x2(const bf16_t* p) {
  return *(const uint32_t*)p;
}

// Async copy 16B/lane from global (saddr + 32-bit voffset) into LDS.
// ISA: GLOBAL_LOAD_ASYNC_TO_LDS_B128 (tracked by ASYNCcnt).
static __device__ __forceinline__ void async_ld_b128(uint32_t lds_off,
                                                     uint32_t voff,
                                                     const void* base) {
  asm volatile("global_load_async_to_lds_b128 %0, %1, %2"
               :
               : "v"(lds_off), "v"(voff),
                 "s"((unsigned long long)(uintptr_t)base)
               : "memory");
}
static __device__ __forceinline__ void wait_asynccnt0() {
  asm volatile("s_wait_asynccnt 0" ::: "memory");
}

// DPP16 row_ror:<N> move (operates within 16-lane rows, matching the
// C-fragment half-group row mapping).
template <int CTRL>
static __device__ __forceinline__ float dpp_rot(float x) {
  return __int_as_float(__builtin_amdgcn_update_dpp(
      0, __float_as_int(x), CTRL, 0xF, 0xF, true));
}

// Row reductions across the 16-lane half-groups via DPP rotations
// (valid for commutative ops; every lane ends with the full row value).
static __device__ __forceinline__ float rowmax16(float v) {
  v = fmaxf(v, dpp_rot<0x128>(v));  // row_ror:8
  v = fmaxf(v, dpp_rot<0x124>(v));  // row_ror:4
  v = fmaxf(v, dpp_rot<0x122>(v));  // row_ror:2
  v = fmaxf(v, dpp_rot<0x121>(v));  // row_ror:1
  return v;
}
static __device__ __forceinline__ float rowsum16(float v) {
  v += dpp_rot<0x128>(v);
  v += dpp_rot<0x124>(v);
  v += dpp_rot<0x122>(v);
  v += dpp_rot<0x121>(v);
  return v;
}

// A fragment, 16x32 bf16, row-major source with leading dim ld (elements).
// Contiguous k order: lane<16 holds row=lane, K=0..15; lane>=16: K=16..31
static __device__ __forceinline__ v16bf load_a_frag(const bf16_t* p, int ld) {
  const int lane = threadIdx.x & 31;
  const int row  = lane & 15;
  const int kb   = (lane >> 4) * 16;
  const bf16_t* r = p + (size_t)row * ld;
  v16bf f;
#pragma unroll
  for (int i = 0; i < 8; ++i) f[i] = r[kb + i];
#pragma unroll
  for (int i = 0; i < 8; ++i) f[8 + i] = r[kb + 8 + i];
  return f;
}

// B fragment, 32x16 bf16, from an N-major buffer bn[n][k] (row n = out col).
// Contiguous k order (as A): lane<16 -> n=lane, K=0..15 ; lane>=16 -> n=lane-16, K=16..31
static __device__ __forceinline__ v16bf load_bt_frag(const bf16_t* p, int ld) {
  const int lane = threadIdx.x & 31;
  const int n    = lane & 15;
  const int kb   = (lane >> 4) * 16;
  const bf16_t* r = p + (size_t)n * ld + kb;
  v16bf f;
#pragma unroll
  for (int i = 0; i < 16; ++i) f[i] = r[i];
  return f;
}

// B fragment, 32x16 bf16, from a K-major buffer bk[k][n] (V tiles).
static __device__ __forceinline__ v16bf load_bk_frag(const bf16_t* p, int ld) {
  const int lane = threadIdx.x & 31;
  const int n    = lane & 15;
  const int kb   = (lane >> 4) * 16;
  const bf16_t* c = p + (size_t)kb * ld + n;
  v16bf f;
#pragma unroll
  for (int i = 0; i < 16; ++i) f[i] = c[(size_t)i * ld];
  return f;
}

// ---------------------------------------------------------------------------
// GEMM: Out[M,N] = A[M,K] * W[N,K]^T (+ bias). 64x64 block tile, 4 waves,
// 32x32 per wave, K-step 32, packed bf16 LDS staging.
// ---------------------------------------------------------------------------
template <typename AT, typename OT, bool BIAS>
__global__ __launch_bounds__(128) void gemm_nt_wmma(
    const AT* __restrict__ A, const float* __restrict__ W,
    const float* __restrict__ bias, OT* __restrict__ Out,
    int M, int N, int K) {
  constexpr int LDA = 40;  // padded bf16 stride (80B, 16B multiple)
  __shared__ __align__(16) bf16_t As[64][LDA];
  __shared__ __align__(16) bf16_t Bs[64][LDA];

  const int tid  = threadIdx.x;
  const int wave = tid >> 5, lane = tid & 31;
  const int m0 = blockIdx.y * 64, n0 = blockIdx.x * 64;
  const int wr = (wave >> 1) * 32, wc = (wave & 1) * 32;

  v8f acc[2][2];
#pragma unroll
  for (int i = 0; i < 2; ++i)
#pragma unroll
    for (int j = 0; j < 2; ++j) acc[i][j] = {};

  for (int k0 = 0; k0 < K; k0 += 32) {
    __syncthreads();
#pragma unroll
    for (int e = 0; e < 8; ++e) {  // 64 rows x 16 bf16-pairs, 128 threads
      int lin = tid + e * 128;
      int r = lin >> 4, cp = (lin & 15) * 2;
      *(uint32_t*)&As[r][cp] =
          load_pair_as_bf16x2(A + (size_t)(m0 + r) * K + k0 + cp);
      *(uint32_t*)&Bs[r][cp] =
          load_pair_as_bf16x2(W + (size_t)(n0 + r) * K + k0 + cp);
    }
    if (k0 + 32 < K) {  // prefetch next K tile (global_prefetch_b8)
      int pr = tid >> 1, pc = (tid & 1) * 16;
      __builtin_prefetch(A + (size_t)(m0 + pr) * K + k0 + 32 + pc, 0, 3);
      __builtin_prefetch(W + (size_t)(n0 + pr) * K + k0 + 32 + pc, 0, 3);
    }
    __syncthreads();

    v16bf a0 = load_a_frag(&As[wr][0], LDA);
    v16bf a1 = load_a_frag(&As[wr + 16][0], LDA);
    v16bf b0 = load_bt_frag(&Bs[wc][0], LDA);
    v16bf b1 = load_bt_frag(&Bs[wc + 16][0], LDA);
    acc[0][0] = wmma_bf16(a0, b0, acc[0][0]);
    acc[0][1] = wmma_bf16(a0, b1, acc[0][1]);
    acc[1][0] = wmma_bf16(a1, b0, acc[1][0]);
    acc[1][1] = wmma_bf16(a1, b1, acc[1][1]);
  }

  const int rhi = (lane >> 4) << 3, col = lane & 15;
#pragma unroll
  for (int i = 0; i < 2; ++i)
#pragma unroll
    for (int j = 0; j < 2; ++j) {
      const int cb = n0 + wc + j * 16 + col;
      const float bv = BIAS ? bias[cb] : 0.0f;
#pragma unroll
      for (int r = 0; r < 8; ++r) {
        Out[(size_t)(m0 + wr + i * 16 + r + rhi) * N + cb] =
            (OT)(acc[i][j][r] + bv);
      }
    }
}

// ---------------------------------------------------------------------------
// Causal GQA flash attention. Grid: (Hq=32, T/64). Block: 128 threads = 4
// waves; wave w owns query rows [t0+16w, t0+16w+16) x D=64. 32-key steps.
// K/V tiles staged with GLOBAL_LOAD_ASYNC_TO_LDS_B128.
// ---------------------------------------------------------------------------
__global__ __launch_bounds__(128) void gqa_flash_wmma(
    const bf16_t* __restrict__ Q, const bf16_t* __restrict__ Kc,
    const bf16_t* __restrict__ Vc, bf16_t* __restrict__ AO, int T) {
  constexpr int Cq = 2048, Ckv = 512, D = 64;
  constexpr int KLD = 72;  // bf16 stride for 32x64 K/V tiles (144B, 16B mult)
  constexpr int PLD = 40;  // bf16 stride for 16x32 P tiles (80B, 16B mult)
  __shared__ __align__(16) bf16_t Ks[32][KLD];
  __shared__ __align__(16) bf16_t Vs[32][KLD];
  __shared__ __align__(16) bf16_t Ps[4][16][PLD];

  const int tid = threadIdx.x, wave = tid >> 5, lane = tid & 31;
  const int hq = blockIdx.x;       // query head
  const int hk = hq >> 2;          // kv head (G = 4)
  const int t0 = blockIdx.y * 64;  // query block base
  const int m0 = t0 + wave * 16;   // this wave's query rows
  const float scale = 0.125f;      // 1/sqrt(64)

  // Q fragments (A layout), resident for the whole pass.
  const bf16_t* qp = Q + (size_t)m0 * Cq + hq * D;
  v16bf q0 = load_a_frag(qp, Cq);
  v16bf q1 = load_a_frag(qp + 32, Cq);

  v8f o[4];
#pragma unroll
  for (int f = 0; f < 4; ++f) o[f] = {};
  float rmax[8], rsum[8];
#pragma unroll
  for (int r = 0; r < 8; ++r) { rmax[r] = -__builtin_inff(); rsum[r] = 0.0f; }

  const int rhi = (lane >> 4) << 3, cn = lane & 15;
  const int nsteps = t0 / 32 + 2;  // causal: keys up to t0+63

  // Per-wave async staging geometry: wave stages rows [8w, 8w+8) of the
  // 32x64 bf16 tile; each B128 op covers 4 rows (8 lanes x 16B per row).
  const int srow = wave * 8 + (lane >> 3);  // + {0,4} per issue
  const int scb  = (lane & 7) * 16;         // byte column within row

  for (int j = 0; j < nsteps; ++j) {
    const int kb0 = j * 32;
    __syncthreads();  // previous iteration's LDS reads complete
#pragma unroll
    for (int i = 0; i < 2; ++i) {
      const int row = srow + i * 4;
      const uint32_t voff =
          (uint32_t)((kb0 + row) * (Ckv * 2) + hk * (D * 2) + scb);
      async_ld_b128((uint32_t)(size_t)&Ks[row][0] + scb, voff, Kc);
      async_ld_b128((uint32_t)(size_t)&Vs[row][0] + scb, voff, Vc);
    }
    if (j + 1 < nsteps) {  // prefetch next tile's rows
      const int prow = kb0 + 32 + (tid & 31);
      __builtin_prefetch(Kc + (size_t)prow * Ckv + hk * D, 0, 3);
      __builtin_prefetch(Vc + (size_t)prow * Ckv + hk * D, 0, 3);
    }
    wait_asynccnt0();
    __syncthreads();

    // S = Q * K^T over two 16-key tiles (contract D=64 as 2 chained WMMAs)
    float sv[2][8];
#pragma unroll
    for (int kt = 0; kt < 2; ++kt) {
      v16bf b0 = load_bt_frag(&Ks[kt * 16][0], KLD);   // d = 0..31
      v16bf b1 = load_bt_frag(&Ks[kt * 16][32], KLD);  // d = 32..63
      v8f s = {};
      s = wmma_bf16(q0, b0, s);
      s = wmma_bf16(q1, b1, s);
#pragma unroll
      for (int r = 0; r < 8; ++r) sv[kt][r] = s[r] * scale;
    }
    if (kb0 + 31 > m0) {  // wave-uniform: only diagonal blocks need masking
#pragma unroll
      for (int kt = 0; kt < 2; ++kt)
#pragma unroll
        for (int r = 0; r < 8; ++r) {
          const int row  = m0 + r + rhi;
          const int colk = kb0 + kt * 16 + cn;
          if (colk > row) sv[kt][r] = -__builtin_inff();
        }
    }

    // Online softmax (per-row stats via DPP16 row reductions)
#pragma unroll
    for (int r = 0; r < 8; ++r) {
      const float m  = rowmax16(fmaxf(sv[0][r], sv[1][r]));
      const float nm = fmaxf(rmax[r], m);
      const float alpha = __expf(rmax[r] - nm);
      const float p0 = __expf(sv[0][r] - nm);
      const float p1 = __expf(sv[1][r] - nm);
      sv[0][r] = p0; sv[1][r] = p1;
      rsum[r] = rsum[r] * alpha + rowsum16(p0 + p1);
      rmax[r] = nm;
#pragma unroll
      for (int f = 0; f < 4; ++f) o[f][r] *= alpha;
    }

    // P: C layout -> LDS -> A layout
#pragma unroll
    for (int kt = 0; kt < 2; ++kt)
#pragma unroll
      for (int r = 0; r < 8; ++r)
        Ps[wave][r + rhi][kt * 16 + cn] = (bf16_t)sv[kt][r];
    __syncthreads();

    v16bf pa = load_a_frag(&Ps[wave][0][0], PLD);
#pragma unroll
    for (int f = 0; f < 4; ++f) {
      v16bf bv = load_bk_frag(&Vs[0][f * 16], KLD);  // k = key, n = d
      o[f] = wmma_bf16(pa, bv, o[f]);
    }
  }

  // Normalize and emit bf16 for the final projection GEMM.
#pragma unroll
  for (int f = 0; f < 4; ++f)
#pragma unroll
    for (int r = 0; r < 8; ++r)
      AO[(size_t)(m0 + r + rhi) * Cq + hq * D + f * 16 + cn] =
          (bf16_t)(o[f][r] / rsum[r]);
}

// ---------------------------------------------------------------------------
extern "C" void kernel_launch(void* const* d_in, const int* in_sizes, int n_in,
                              void* d_out, int out_size, void* d_ws,
                              size_t ws_size, hipStream_t stream) {
  (void)in_sizes; (void)n_in; (void)out_size; (void)ws_size;
  const float* x  = (const float*)d_in[0];  // [T, C]
  const float* Wq = (const float*)d_in[1];  // [C, C]
  const float* Wk = (const float*)d_in[2];  // [Ckv, C]
  const float* Wv = (const float*)d_in[3];  // [Ckv, C]
  const float* Wo = (const float*)d_in[4];  // [C, C]
  const float* bo = (const float*)d_in[5];  // [C]

  const int T = 2048, Cq = 2048, Ckv = 512, Hq = 32;

  // Workspace: bf16 Q, K, V, attention-out (20 MB total)
  bf16_t* Qb = (bf16_t*)d_ws;
  bf16_t* Kb = Qb + (size_t)T * Cq;
  bf16_t* Vb = Kb + (size_t)T * Ckv;
  bf16_t* AO = Vb + (size_t)T * Ckv;

  dim3 blk(128);
  gemm_nt_wmma<float, bf16_t, false>
      <<<dim3(Cq / 64, T / 64), blk, 0, stream>>>(x, Wq, nullptr, Qb, T, Cq, Cq);
  gemm_nt_wmma<float, bf16_t, false>
      <<<dim3(Ckv / 64, T / 64), blk, 0, stream>>>(x, Wk, nullptr, Kb, T, Ckv, Cq);
  gemm_nt_wmma<float, bf16_t, false>
      <<<dim3(Ckv / 64, T / 64), blk, 0, stream>>>(x, Wv, nullptr, Vb, T, Ckv, Cq);

  gqa_flash_wmma<<<dim3(Hq, T / 64), blk, 0, stream>>>(Qb, Kb, Vb, AO, T);

  gemm_nt_wmma<bf16_t, float, true>
      <<<dim3(Cq / 64, T / 64), blk, 0, stream>>>(AO, Wo, bo, (float*)d_out,
                                                  T, Cq, Cq);
}
